// SingleHeadAttention_71347996721687
// MI455X (gfx1250) — matrix operand order repaired and checked by reading, not compiled, not run on hardware
//
#include <hip/hip_runtime.h>
#include <math.h>

// ---------------------------------------------------------------------------
// Problem constants (reference: B=4, S=4096, E=2048, A=2048)
// ---------------------------------------------------------------------------
#define BATCH 4
#define SEQ   4096
#define EMB   2048
#define ADIM  2048

// LDS B-tile: 64 columns x 32 k-values, row padded to 40 elems (80B) to spread
// banks for conflict-free ds_load_b128 of fragments.
#define BROWS       64
#define BSTRIDE     40
#define BTILE_ELEMS (BROWS * BSTRIDE)

typedef __attribute__((ext_vector_type(16))) __bf16 v16bf;
typedef __attribute__((ext_vector_type(8)))  float  v8f;

union FragBF16 {
    v16bf v;
    uint4 u[2];
};

// ---------------------------------------------------------------------------
// gfx1250 async global->LDS copy of 16B per lane (ASYNCcnt-tracked), and the
// matching wait. LDS destination is passed as a raw 32-bit LDS byte offset.
// ---------------------------------------------------------------------------
__device__ inline void async_copy_b128(__bf16* lds_dst, const __bf16* g_src)
{
    unsigned loff = (unsigned)(unsigned long long)
        (__attribute__((address_space(3))) void*)lds_dst;
    asm volatile("global_load_async_to_lds_b128 %0, %1, off"
                 :: "v"(loff), "v"(g_src)
                 : "memory");
}

__device__ inline void wait_async0()
{
    asm volatile("s_wait_asynccnt 0x0" ::: "memory");
}

// ---------------------------------------------------------------------------
// Block-level GEMM: 256 threads (8 waves). Block computes 128(M) x 64(N) of
//   C = A * B^T
//   A: row-major [M x K], lda (contraction contiguous)  - per-wave global b128
//   B: row-major [N x K], ldb (contraction contiguous)  - staged in LDS via
//      double-buffered global_load_async_to_lds_b128, shared by all 8 waves.
// Each wave owns one 16-row strip and 4 WMMA accumulators (16x64).
//
// Fragment layouts per CDNA5 ISA:
//   16-bit A 16x32: lanes 0-15: M=lane, K=k0+0..7 / +16..23
//                   lanes 16-31: M=lane-16, K=k0+8..15 / +24..31
//   16-bit B 32x16: lanes 0-15: N=lane, K=k0+0..7 / +16..23; lanes 16-31: K=k0+8..15 / +24..31
// ---------------------------------------------------------------------------
__device__ inline void block_gemm_128x64(const __bf16* __restrict__ Aptr,
                                         const __bf16* __restrict__ Bptr,
                                         int lda, int ldb,
                                         int mBlock, int n0,
                                         int kEnd,            // block-uniform, mult of 32
                                         v8f acc[4],
                                         __bf16* ldsB)        // [2][BTILE_ELEMS]
{
    const int tid  = threadIdx.x;        // 0..255
    const int lane = tid & 31;
    const int wave = tid >> 5;
    const int m0   = mBlock + wave * 16;

    // A fragment addressing (direct global)
    const int arow = m0 + (lane & 15);
    const int aoff = (lane >> 4) * 8;                 // 0 or 8
    const __bf16* __restrict__ Abase = Aptr + (size_t)arow * lda + aoff;

    // B staging addressing: thread t copies 16B: col = t/4, sub-chunk = t%4
    const int scol = tid >> 2;                        // 0..63
    const int ssub = tid & 3;                         // 0..3 (8 elems each)
    const __bf16* __restrict__ Bsrc = Bptr + (size_t)(n0 + scol) * ldb + ssub * 8;
    const int sdst = scol * BSTRIDE + ssub * 8;       // elems within a buffer

    // B fragment read addressing (from LDS)
    const int bcol = lane & 15;
    const int boff = (lane >> 4) * 8;                 // 0 or 8

    // Prefetch first B tile into buffer 0
    async_copy_b128(ldsB + sdst, Bsrc);

    int cur = 0;
    for (int k = 0; k < kEnd; k += 32) {
        wait_async0();          // our prefetched tile has landed
        __syncthreads();        // everyone's chunks landed; prev buffer free

        if (k + 32 < kEnd)      // prefetch next tile into the other buffer
            async_copy_b128(ldsB + (1 - cur) * BTILE_ELEMS + sdst, Bsrc + k + 32);

        // A fragment: two b128 loads
        FragBF16 a;
        a.u[0] = *reinterpret_cast<const uint4*>(Abase + k);
        a.u[1] = *reinterpret_cast<const uint4*>(Abase + k + 16);
        __builtin_prefetch(Abase + k + 64, 0, 1);     // global_prefetch A stream

        const __bf16* Bt = ldsB + cur * BTILE_ELEMS;
#pragma unroll
        for (int t = 0; t < 4; ++t) {
            const __bf16* p = Bt + (t * 16 + bcol) * BSTRIDE + boff;
            FragBF16 b;
            b.u[0] = *reinterpret_cast<const uint4*>(p);
            b.u[1] = *reinterpret_cast<const uint4*>(p + 16);
            acc[t] = __builtin_amdgcn_wmma_f32_16x16x32_bf16(
                false, a.v, false, b.v, (short)0, acc[t], false, false);
        }
        cur ^= 1;
    }
}

// ---------------------------------------------------------------------------
// Kernel 0: fp32 -> bf16 cast with scale (folds 1/sqrt(A) into Wq)
// ---------------------------------------------------------------------------
__global__ void cast_f32_to_bf16(const float* __restrict__ x,
                                 __bf16* __restrict__ y,
                                 size_t n, float scale)
{
    size_t i      = (size_t)blockIdx.x * blockDim.x + threadIdx.x;
    size_t stride = (size_t)gridDim.x * blockDim.x;
    for (; i < n; i += stride)
        y[i] = (__bf16)(x[i] * scale);
}

// ---------------------------------------------------------------------------
// Kernel 1: projection GEMM  Y = X * W^T
//   transposed==0: Y row-major [M, A] (Q, K)
//   transposed==1: Y per-batch transposed [B, A, S] (V^T)
// ---------------------------------------------------------------------------
__global__ void proj_gemm(const __bf16* __restrict__ Xb,
                          const __bf16* __restrict__ Wb,
                          __bf16* __restrict__ Y,
                          int N, int Kdim, int S, int transposed)
{
    __shared__ __bf16 ldsB[2 * BTILE_ELEMS];

    const int mBlock = blockIdx.x * 128;
    const int n0     = blockIdx.y * 64;

    v8f acc[4] = {};
    block_gemm_128x64(Xb, Wb, Kdim, Kdim, mBlock, n0, Kdim, acc, ldsB);

    const int lane = threadIdx.x & 31;
    const int wave = threadIdx.x >> 5;
    const int m0   = mBlock + wave * 16;
    const int crow = (lane >> 4) * 8;
    const int ccol = lane & 15;

    if (!transposed) {
#pragma unroll
        for (int t = 0; t < 4; ++t)
#pragma unroll
            for (int r = 0; r < 8; ++r) {
                int m = m0 + crow + r;
                int n = n0 + t * 16 + ccol;
                Y[(size_t)m * N + n] = (__bf16)acc[t][r];
            }
    } else {
#pragma unroll
        for (int t = 0; t < 4; ++t)
#pragma unroll
            for (int r = 0; r < 8; ++r) {
                int m = m0 + crow + r;
                int b = m / S;
                int s = m - b * S;
                int n = n0 + t * 16 + ccol;
                Y[((size_t)b * N + n) * S + s] = (__bf16)acc[t][r];
            }
    }
}

// ---------------------------------------------------------------------------
// Kernel 2: scores = Q K^T (per batch), bf16 out. Fully-masked causal blocks
// skip (block-uniform); softmax never reads k > q. Scale folded into Q.
// ---------------------------------------------------------------------------
__global__ void scores_gemm(const __bf16* __restrict__ Qb,
                            const __bf16* __restrict__ Kb,
                            __bf16* __restrict__ Sc)
{
    const int b      = blockIdx.z;
    const int mBlock = blockIdx.x * 128;              // query block
    const int n0     = blockIdx.y * 64;               // key block
    if (n0 > mBlock + 127) return;                    // block-uniform early out

    __shared__ __bf16 ldsB[2 * BTILE_ELEMS];

    const __bf16* Q = Qb + (size_t)b * SEQ * ADIM;
    const __bf16* K = Kb + (size_t)b * SEQ * ADIM;
    __bf16*       S = Sc + (size_t)b * SEQ * SEQ;

    v8f acc[4] = {};
    block_gemm_128x64(Q, K, ADIM, ADIM, mBlock, n0, ADIM, acc, ldsB);

    const int lane = threadIdx.x & 31;
    const int wave = threadIdx.x >> 5;
    const int m0   = mBlock + wave * 16;
    const int crow = (lane >> 4) * 8;
    const int ccol = lane & 15;
#pragma unroll
    for (int t = 0; t < 4; ++t)
#pragma unroll
        for (int r = 0; r < 8; ++r) {
            int q = m0 + crow + r;
            int k = n0 + t * 16 + ccol;
            S[(size_t)q * SEQ + k] = (__bf16)acc[t][r];
        }
}

// ---------------------------------------------------------------------------
// Kernel 3: causal row softmax. One 256-thread block per (b, q) row.
// Reads only k <= q; writes zeros for k > q so PV can run 32-padded K ranges.
// ---------------------------------------------------------------------------
__global__ void softmax_rows(const __bf16* __restrict__ Sc,
                             __bf16* __restrict__ Pr)
{
    const int q = blockIdx.x;
    const int b = blockIdx.y;
    const __bf16* row  = Sc + ((size_t)b * SEQ + q) * SEQ;
    __bf16*       prow = Pr + ((size_t)b * SEQ + q) * SEQ;

    const int tid  = threadIdx.x;
    const int nvis = q + 1;

    __shared__ float red[256];

    float lmax = -3.0e38f;
    for (int k = tid; k < nvis; k += 256)
        lmax = fmaxf(lmax, (float)row[k]);
    red[tid] = lmax;
    __syncthreads();
    for (int s = 128; s > 0; s >>= 1) {
        if (tid < s) red[tid] = fmaxf(red[tid], red[tid + s]);
        __syncthreads();
    }
    const float m = red[0];
    __syncthreads();

    float lsum = 0.0f;
    for (int k = tid; k < nvis; k += 256)
        lsum += __expf((float)row[k] - m);
    red[tid] = lsum;
    __syncthreads();
    for (int s = 128; s > 0; s >>= 1) {
        if (tid < s) red[tid] += red[tid + s];
        __syncthreads();
    }
    const float inv = 1.0f / red[0];

    for (int k = tid; k < SEQ; k += 256) {
        float p = (k < nvis) ? __expf((float)row[k] - m) * inv : 0.0f;
        prow[k] = (__bf16)p;
    }
}

// ---------------------------------------------------------------------------
// Kernel 4: out = P * V  (P [S,S] bf16, V^T [A,S] bf16, per batch).
// K-loop clipped to block-uniform causal limit (probs zero-padded past q).
// fp32 out with reference's round-to-4-decimals.
// ---------------------------------------------------------------------------
__global__ void pv_gemm(const __bf16* __restrict__ Pr,
                        const __bf16* __restrict__ Vt,
                        float* __restrict__ Out)
{
    __shared__ __bf16 ldsB[2 * BTILE_ELEMS];

    const int b      = blockIdx.z;
    const int mBlock = blockIdx.x * 128;              // query block
    const int n0     = blockIdx.y * 64;               // output-dim block

    int kEnd = mBlock + 128;                          // causal limit (mult of 32)
    if (kEnd > SEQ) kEnd = SEQ;

    const __bf16* P = Pr + (size_t)b * SEQ * SEQ;
    const __bf16* V = Vt + (size_t)b * ADIM * SEQ;

    v8f acc[4] = {};
    block_gemm_128x64(P, V, SEQ, SEQ, mBlock, n0, kEnd, acc, ldsB);

    const int lane = threadIdx.x & 31;
    const int wave = threadIdx.x >> 5;
    const int m0   = mBlock + wave * 16;
    const int crow = (lane >> 4) * 8;
    const int ccol = lane & 15;
#pragma unroll
    for (int t = 0; t < 4; ++t)
#pragma unroll
        for (int r = 0; r < 8; ++r) {
            int q = m0 + crow + r;
            int a = n0 + t * 16 + ccol;
            float v = acc[t][r];
            v = rintf(v * 1.0e4f) * 1.0e-4f;          // torch.round(..., 4)
            Out[((size_t)b * SEQ + q) * ADIM + a] = v;
        }
}

// ---------------------------------------------------------------------------
// Host launcher
// ---------------------------------------------------------------------------
extern "C" void kernel_launch(void* const* d_in, const int* in_sizes, int n_in,
                              void* d_out, int out_size, void* d_ws, size_t ws_size,
                              hipStream_t stream)
{
    const float* embedded = (const float*)d_in[0];  // [B,S,E]
    const float* Wk       = (const float*)d_in[1];  // [A,E]
    const float* Wq       = (const float*)d_in[2];  // [A,E]
    const float* Wv       = (const float*)d_in[3];  // [A,E]
    float*       out      = (float*)d_out;          // [B,S,A]

    const size_t nX = (size_t)BATCH * SEQ * EMB;    // 33,554,432
    const size_t nW = (size_t)ADIM * EMB;           //  4,194,304
    const size_t nQ = (size_t)BATCH * SEQ * ADIM;   // 33,554,432
    const size_t nS = (size_t)BATCH * SEQ * SEQ;    // 67,108,864

    // Workspace layout (bf16 elements), total 536 MB
    char* ws = (char*)d_ws;
    __bf16* Xb  = (__bf16*)(ws);                                   // 64 MB
    __bf16* Wkb = (__bf16*)(ws + nX * 2);                          //  8 MB
    __bf16* Wqb = Wkb + nW;                                        //  8 MB
    __bf16* Wvb = Wqb + nW;                                        //  8 MB
    __bf16* Qb  = Wvb + nW;                                        // 64 MB
    __bf16* Kb  = Qb + nQ;                                         // 64 MB
    __bf16* Vt  = Kb + nQ;                                         // 64 MB
    __bf16* Sc  = Vt + nQ;                                         // 128 MB
    __bf16* Pr  = Sc + nS;                                         // 128 MB

    const float qscale = 1.0f / sqrtf((float)ADIM);

    // ---- casts to bf16 (scale folded into Wq) ----
    cast_f32_to_bf16<<<8192, 256, 0, stream>>>(embedded, Xb, nX, 1.0f);
    cast_f32_to_bf16<<<4096, 256, 0, stream>>>(Wk, Wkb, nW, 1.0f);
    cast_f32_to_bf16<<<4096, 256, 0, stream>>>(Wq, Wqb, nW, qscale);
    cast_f32_to_bf16<<<4096, 256, 0, stream>>>(Wv, Wvb, nW, 1.0f);

    // ---- QKV projections: [16384,2048] x [2048,2048]^T ----
    {
        dim3 grid((BATCH * SEQ) / 128, ADIM / 64, 1);
        proj_gemm<<<grid, 256, 0, stream>>>(Xb, Wqb, Qb, ADIM, EMB, SEQ, 0);
        proj_gemm<<<grid, 256, 0, stream>>>(Xb, Wkb, Kb, ADIM, EMB, SEQ, 0);
        proj_gemm<<<grid, 256, 0, stream>>>(Xb, Wvb, Vt, ADIM, EMB, SEQ, 1); // V^T
    }

    // ---- scores = Q K^T (causal blocks only) ----
    {
        dim3 grid(SEQ / 128, SEQ / 64, BATCH);
        scores_gemm<<<grid, 256, 0, stream>>>(Qb, Kb, Sc);
    }

    // ---- causal softmax rows ----
    {
        dim3 grid(SEQ, BATCH, 1);
        softmax_rows<<<grid, 256, 0, stream>>>(Sc, Pr);
    }

    // ---- out = P V ----
    {
        dim3 grid(SEQ / 128, ADIM / 64, BATCH);
        pv_gemm<<<grid, 256, 0, stream>>>(Pr, Vt, out);
    }
}
